// AdaptiveExpertLayer_8641474200507
// MI455X (gfx1250) — matrix operand order repaired and checked by reading, not compiled, not run on hardware
//
#include <hip/hip_runtime.h>
#include <hip/hip_bf16.h>

#define D_MODEL 1024
#define D_FF    4096
#define NEXP    8
#define NTOK    8192   // B*S = 4*2048
#define MT      32     // token tile (M)
#define FFC     512    // ff chunk held in LDS per iteration
#define XP      1040   // X tile row pitch (elements), 32B-aligned rows
#define HP      528    // H chunk row pitch (elements), 32B-aligned rows
#define THREADS 256    // 8 waves (wave32)

typedef __attribute__((ext_vector_type(16))) __bf16 v16bf;
typedef __attribute__((ext_vector_type(8)))  __bf16 v8bf;
typedef __attribute__((ext_vector_type(8)))  float  v8f;

// ---------------- workspace layout (bytes) ----------------
#define WS_COUNTS 0                          // int[NEXP], padded
#define WS_IDX    256                        // int[2*NTOK]
#define WS_WTOP   (WS_IDX  + 2*NTOK*4)       // float[2*NTOK]
#define WS_LTOK   (WS_WTOP + 2*NTOK*4)       // int[NEXP*NTOK]
#define WS_LW     (WS_LTOK + NEXP*NTOK*4)    // float[NEXP*NTOK]
// total ~0.66 MB

#define SMEM_HDR   256
#define SMEM_BYTES (SMEM_HDR + MT*XP*2 + MT*HP*2)   // ~100.6 KB of 320 KB/WGP

// ---------------------------------------------------------------------------
// Kernel 1: router — logits, softmax, top-2, renormalize.
// ---------------------------------------------------------------------------
__global__ void moe_router(const float* __restrict__ x,
                           const float* __restrict__ gw,
                           int* __restrict__ idx2,
                           float* __restrict__ w2) {
    int t = blockIdx.x * blockDim.x + threadIdx.x;
    if (t >= NTOK) return;
    const float* xr = x + (size_t)t * D_MODEL;
    float acc[NEXP];
#pragma unroll
    for (int e = 0; e < NEXP; ++e) acc[e] = 0.f;
    for (int d = 0; d < D_MODEL; d += 4) {
        float x0 = xr[d], x1 = xr[d + 1], x2 = xr[d + 2], x3 = xr[d + 3];
#pragma unroll
        for (int e = 0; e < NEXP; ++e) {
            const float* g = gw + e * D_MODEL + d;
            acc[e] += x0 * g[0] + x1 * g[1] + x2 * g[2] + x3 * g[3];
        }
    }
    float mx = acc[0];
#pragma unroll
    for (int e = 1; e < NEXP; ++e) mx = fmaxf(mx, acc[e]);
    float p[NEXP], sum = 0.f;
#pragma unroll
    for (int e = 0; e < NEXP; ++e) { p[e] = __expf(acc[e] - mx); sum += p[e]; }
    // top-2, ties -> lowest index (strict >)
    int i0 = 0; float b0 = p[0];
#pragma unroll
    for (int e = 1; e < NEXP; ++e) if (p[e] > b0) { b0 = p[e]; i0 = e; }
    int i1 = (i0 == 0) ? 1 : 0; float b1v = p[i1];
#pragma unroll
    for (int e = 0; e < NEXP; ++e)
        if (e != i0 && p[e] > b1v) { b1v = p[e]; i1 = e; }
    float pr0 = b0 / sum, pr1 = b1v / sum;
    float inv = 1.f / (pr0 + pr1 + 1e-9f);
    idx2[2 * t] = i0;  idx2[2 * t + 1] = i1;
    w2[2 * t] = pr0 * inv;  w2[2 * t + 1] = pr1 * inv;
}

// ---------------------------------------------------------------------------
// Kernel 2: deterministic per-expert token lists via LDS prefix scan.
// One block per expert; token order preserved -> bit-identical every call.
// ---------------------------------------------------------------------------
__global__ void moe_build_lists(const int* __restrict__ idx2,
                                const float* __restrict__ w2,
                                int* __restrict__ counts,
                                int* __restrict__ ltok,
                                float* __restrict__ lwt) {
    const int e = blockIdx.x;
    const int tid = threadIdx.x;
    __shared__ int sc[256];
    int base = 0;
    for (int c = 0; c < NTOK / 256; ++c) {
        int t = c * 256 + tid;
        int i0 = idx2[2 * t], i1 = idx2[2 * t + 1];
        int f = (i0 == e || i1 == e) ? 1 : 0;
        float w = (i0 == e) ? w2[2 * t] : w2[2 * t + 1];
        sc[tid] = f;
        __syncthreads();
#pragma unroll
        for (int off = 1; off < 256; off <<= 1) {
            int v = (tid >= off) ? sc[tid - off] : 0;
            __syncthreads();
            sc[tid] += v;
            __syncthreads();
        }
        if (f) {
            int slot = base + sc[tid] - 1;
            ltok[e * NTOK + slot] = t;
            lwt[e * NTOK + slot]  = w;
        }
        int tot = sc[255];
        __syncthreads();
        base += tot;
    }
    if (tid == 0) counts[e] = base;
}

// ---------------------------------------------------------------------------
// Fragment helpers (contiguous k = 16*(L>>4) + j order, same for A and B, wave32)
// ---------------------------------------------------------------------------
// A (16x32 bf16): lane L: M = L&15; K = (L>>4)*16 + j.
__device__ __forceinline__ v16bf lds_a_frag(const __bf16* row, int kb, int lhi) {
    v8bf lo = *(const v8bf*)(row + kb + lhi * 16);
    v8bf hi = *(const v8bf*)(row + kb + 8 + lhi * 16);
    v16bf a;
#pragma unroll
    for (int j = 0; j < 8; ++j) { a[j] = lo[j]; a[j + 8] = hi[j]; }
    return a;
}
// B (32x16 bf16): lane L: N = L&15; K = (L>>4)*16 + j  -> 16 contiguous f32
// from the weight row (weights are [out, in] so W^T columns are W rows).
__device__ __forceinline__ v16bf ld_b_frag_cvt(const float* __restrict__ p) {
    v16bf b;
#pragma unroll
    for (int j = 0; j < 16; ++j) b[j] = (__bf16)p[j];
    return b;
}

// ---------------------------------------------------------------------------
// Kernel 3: fused expert MLP. Grid (NTOK/MT tiles, NEXP experts).
// Stage A: h = relu(X W1^T + b1) chunkwise into LDS (bf16).
// Stage B: y accumulators (f32, VGPR-resident) += h W2^T.
// Epilogue: out[token] += gatew * (y + b2)  (2 contributions/elem, order-safe).
// ---------------------------------------------------------------------------
__global__ void moe_expert_fused(const float* __restrict__ x,
                                 const float* __restrict__ W1,
                                 const float* __restrict__ b1,
                                 const float* __restrict__ W2,
                                 const float* __restrict__ b2,
                                 const int* __restrict__ counts,
                                 const int* __restrict__ ltok,
                                 const float* __restrict__ lwt,
                                 float* __restrict__ out) {
    extern __shared__ char smem[];
    int*    tok = (int*)smem;                         // [MT]
    float*  twt = (float*)(smem + MT * 4);            // [MT]
    __bf16* Xs  = (__bf16*)(smem + SMEM_HDR);         // [MT][XP]
    __bf16* Hs  = (__bf16*)(smem + SMEM_HDR + MT * XP * 2); // [MT][HP]

    const int e    = blockIdx.y;
    const int tile = blockIdx.x;
    const int cnt  = counts[e];
    if (tile * MT >= cnt) return;

    const int tid  = threadIdx.x;
    const int lane = tid & 31;
    const int wave = tid >> 5;
    const int llo  = lane & 15;
    const int lhi  = lane >> 4;

    if (tid < MT) {
        int g = tile * MT + tid;
        tok[tid] = (g < cnt) ? ltok[e * NTOK + g] : 0;
        twt[tid] = (g < cnt) ? lwt[e * NTOK + g] : 0.f;
    }
    __syncthreads();

    // gather + f32->bf16 convert the X tile (coalesced along d)
    for (int i = tid; i < MT * D_MODEL; i += THREADS) {
        int r = i >> 10;
        int d = i & (D_MODEL - 1);
        Xs[r * XP + d] = (__bf16)x[(size_t)tok[r] * D_MODEL + d];
    }
    __syncthreads();

    const float* W1e = W1 + (size_t)e * D_FF * D_MODEL;
    const float* W2e = W2 + (size_t)e * D_MODEL * D_FF;

    const v8f vzero = {0.f, 0.f, 0.f, 0.f, 0.f, 0.f, 0.f, 0.f};
    v8f acc[16];
#pragma unroll
    for (int i = 0; i < 16; ++i) acc[i] = vzero;

    for (int fc = 0; fc < D_FF / FFC; ++fc) {
        // ---- Stage A: h[:, fc*FFC .. +FFC) = relu(X W1^T + b1) ----
#pragma unroll
        for (int fti = 0; fti < 4; ++fti) {
            const int ftl   = (wave * 4 + fti) * 16;       // ff base in chunk
            const int nglob = fc * FFC + ftl + llo;        // W1 row (ff idx)
            const float* wrow = W1e + (size_t)nglob * D_MODEL + lhi * 16;
            v8f c0 = vzero, c1 = vzero;
#pragma unroll 2
            for (int kb = 0; kb < D_MODEL; kb += 32) {
                __builtin_prefetch(wrow + kb + 64, 0, 0);
                v16bf bf = ld_b_frag_cvt(wrow + kb);
                v16bf a0 = lds_a_frag(Xs + llo * XP,        kb, lhi);
                v16bf a1 = lds_a_frag(Xs + (llo + 16) * XP, kb, lhi);
                c0 = __builtin_amdgcn_wmma_f32_16x16x32_bf16(
                        false, a0, false, bf, (short)0, c0, false, false);
                c1 = __builtin_amdgcn_wmma_f32_16x16x32_bf16(
                        false, a1, false, bf, (short)0, c1, false, false);
            }
            const float bias = b1[e * D_FF + nglob];
#pragma unroll
            for (int v = 0; v < 8; ++v) {
                int m = v + lhi * 8;                         // C row mapping
                Hs[m * HP + ftl + llo]        = (__bf16)fmaxf(c0[v] + bias, 0.f);
                Hs[(m + 16) * HP + ftl + llo] = (__bf16)fmaxf(c1[v] + bias, 0.f);
            }
        }
        __syncthreads();
        // ---- Stage B: acc += h_chunk W2^T (A frags reused across 8 d-tiles)
#pragma unroll 2
        for (int kb = 0; kb < FFC; kb += 32) {
            v16bf a0 = lds_a_frag(Hs + llo * HP,        kb, lhi);
            v16bf a1 = lds_a_frag(Hs + (llo + 16) * HP, kb, lhi);
#pragma unroll
            for (int i = 0; i < 8; ++i) {
                const int dglob = (wave * 8 + i) * 16 + llo; // W2 row (d idx)
                const float* wrow =
                    W2e + (size_t)dglob * D_FF + fc * FFC + kb + lhi * 16;
                v16bf bf = ld_b_frag_cvt(wrow);
                acc[2 * i] = __builtin_amdgcn_wmma_f32_16x16x32_bf16(
                        false, a0, false, bf, (short)0, acc[2 * i], false, false);
                acc[2 * i + 1] = __builtin_amdgcn_wmma_f32_16x16x32_bf16(
                        false, a1, false, bf, (short)0, acc[2 * i + 1], false, false);
            }
        }
        __syncthreads();
    }

    // ---- Epilogue: out[tok] += w * (y + b2); exactly 2 adds/elem from 0 ----
#pragma unroll
    for (int i = 0; i < 8; ++i) {
        const int dglob = (wave * 8 + i) * 16 + llo;
        const float bias = b2[e * D_MODEL + dglob];
#pragma unroll
        for (int v = 0; v < 8; ++v) {
            int m0 = v + lhi * 8;
            int m1 = m0 + 16;
            if (tile * MT + m0 < cnt)
                unsafeAtomicAdd(out + (size_t)tok[m0] * D_MODEL + dglob,
                                twt[m0] * (acc[2 * i][v] + bias));
            if (tile * MT + m1 < cnt)
                unsafeAtomicAdd(out + (size_t)tok[m1] * D_MODEL + dglob,
                                twt[m1] * (acc[2 * i + 1][v] + bias));
        }
    }
}

// ---------------------------------------------------------------------------
extern "C" void kernel_launch(void* const* d_in, const int* in_sizes, int n_in,
                              void* d_out, int out_size, void* d_ws, size_t ws_size,
                              hipStream_t stream) {
    const float* x    = (const float*)d_in[0];
    const float* gw   = (const float*)d_in[1];
    const float* W1   = (const float*)d_in[2];
    const float* b1   = (const float*)d_in[3];
    const float* W2   = (const float*)d_in[4];
    const float* b2   = (const float*)d_in[5];
    float*       out  = (float*)d_out;
    char*        ws   = (char*)d_ws;

    int*   counts = (int*)(ws + WS_COUNTS);
    int*   idx2   = (int*)(ws + WS_IDX);
    float* wtop   = (float*)(ws + WS_WTOP);
    int*   ltok   = (int*)(ws + WS_LTOK);
    float* lwt    = (float*)(ws + WS_LW);

    hipMemsetAsync(d_out, 0, (size_t)out_size * sizeof(float), stream);

    moe_router<<<NTOK / 256, 256, 0, stream>>>(x, gw, idx2, wtop);
    moe_build_lists<<<NEXP, 256, 0, stream>>>(idx2, wtop, counts, ltok, lwt);

    dim3 grid(NTOK / MT, NEXP);
    moe_expert_fused<<<grid, THREADS, SMEM_BYTES, stream>>>(
        x, W1, b1, W2, b2, counts, ltok, lwt, out);
}
